// Memory_81252191306481
// MI455X (gfx1250) — matrix operand order repaired and checked by reading, not compiled, not run on hardware
//
#include <hip/hip_runtime.h>
#include <hip/hip_bf16.h>

typedef __attribute__((ext_vector_type(16))) __bf16  v16bf;
typedef __attribute__((ext_vector_type(8)))  __bf16  v8bf;
typedef __attribute__((ext_vector_type(8)))  float   v8f;
typedef __attribute__((ext_vector_type(4)))  float   v4f;

#define MMB  512   // min/max reduction blocks
#define CB   512   // main score/topk blocks
#define TILE 64    // memory rows per macro-tile
#define TOPK 5
#define NEG_INF (-3.402823466e38f)

// ---------------------------------------------------------------------------
// Kernel A: grid-stride global min/max of memory -> per-block partials
// ---------------------------------------------------------------------------
__global__ __launch_bounds__(256) void mm_kernel(const float* __restrict__ mem,
                                                 int total,
                                                 float* __restrict__ part) {
  __shared__ float smin[256], smax[256];
  int t = threadIdx.x;
  float mn = 3.4e38f, mx = -3.4e38f;
  for (int i = blockIdx.x * 256 + t; i < total; i += gridDim.x * 256) {
    float v = mem[i];
    mn = fminf(mn, v);
    mx = fmaxf(mx, v);
  }
  smin[t] = mn; smax[t] = mx;
  __syncthreads();
  for (int s = 128; s > 0; s >>= 1) {
    if (t < s) {
      smin[t] = fminf(smin[t], smin[t + s]);
      smax[t] = fmaxf(smax[t], smax[t + s]);
    }
    __syncthreads();
  }
  if (t == 0) { part[blockIdx.x * 2] = smin[0]; part[blockIdx.x * 2 + 1] = smax[0]; }
}

// ---------------------------------------------------------------------------
// Kernel B (1 block): finish min/max -> scale/zp; q2 = (query@Wq.T)@Wk
// ---------------------------------------------------------------------------
__global__ __launch_bounds__(256) void prep_kernel(const float* __restrict__ part, int nparts,
                                                   const float* __restrict__ query,
                                                   const float* __restrict__ Wq,
                                                   const float* __restrict__ Wk,
                                                   float* __restrict__ sczp,
                                                   float* __restrict__ q2f) {
  __shared__ float smin[256], smax[256];
  __shared__ float qld[64 * 64];
  int t = threadIdx.x;
  float mn = 3.4e38f, mx = -3.4e38f;
  for (int i = t; i < nparts; i += 256) {
    mn = fminf(mn, part[2 * i]);
    mx = fmaxf(mx, part[2 * i + 1]);
  }
  smin[t] = mn; smax[t] = mx;
  __syncthreads();
  for (int s = 128; s > 0; s >>= 1) {
    if (t < s) {
      smin[t] = fminf(smin[t], smin[t + s]);
      smax[t] = fmaxf(smax[t], smax[t + s]);
    }
    __syncthreads();
  }
  if (t == 0) {
    float scale = (smax[0] - smin[0]) / 255.0f;
    float zp = -smin[0] / scale;
    sczp[0] = scale; sczp[1] = zp; sczp[2] = 1.0f / scale;
    sczp[3] = -zp * scale;   // c1 : dequant = rint(x*inv+zp)*scale + c1
  }
  // q = query @ Wq.T   (q[b][i] = sum_j query[b][j] * Wq[i][j])
  for (int o = t; o < 4096; o += 256) {
    int b = o >> 6, i = o & 63;
    float acc = 0.f;
    for (int j = 0; j < 64; ++j) acc += query[b * 64 + j] * Wq[i * 64 + j];
    qld[o] = acc;
  }
  __syncthreads();
  // q2 = q @ Wk        (q2[b][d] = sum_i q[b][i] * Wk[i][d])
  for (int o = t; o < 4096; o += 256) {
    int b = o >> 6, d = o & 63;
    float acc = 0.f;
    for (int i = 0; i < 64; ++i) acc += qld[b * 64 + i] * Wk[i * 64 + d];
    q2f[o] = acc;
  }
}

// ---------------------------------------------------------------------------
// Kernel C: stream memory once; dequant -> bf16 LDS macro-tile (64 rows);
//           WMMA with A = mem rows, B = q2 (queries): each lane's
//           accumulator holds 8 scores of ONE query column. Register-only
//           streaming top-5 on all 32 lanes behind a max-of-8 fast gate;
//           half-wave merge via shuffles at the end.
//   block = 128 threads = 4 waves; wave w owns queries 16w..16w+15.
// ---------------------------------------------------------------------------
__global__ __launch_bounds__(128) void score_topk_kernel(
    const float* __restrict__ mem,
    const float* __restrict__ attw,
    const float* __restrict__ sczp,
    const float* __restrict__ q2f,
    float* __restrict__ pscore, int* __restrict__ pidx,
    int N, int chunk) {
  __shared__ __bf16 tileB[TILE * 64];   // 64 mem rows x 64 dims, bf16 (8KB)
  __shared__ float  s_aw[TILE];         // attention bias for this tile

  int t    = threadIdx.x;
  int lane = t & 31;
  int w    = t >> 5;
  int nq   = lane & 15;   // query column owned by this lane (within wave)
  int hi   = lane >> 4;

  float scale = sczp[0], zp = sczp[1], invs = sczp[2], c1 = sczp[3];
  (void)scale;

  // --- B fragments = q2 columns, built once (contiguous k order, same as A:
  //     lane = n + 16*(k/16), element e = k%16) -----------------------------
  v16bf b0, b1;
  {
    const float* qrow = q2f + (w * 16 + nq) * 64;
#pragma unroll
    for (int e = 0; e < 16; ++e) {
      int k = 16 * hi + e;
      b0[e] = (__bf16)qrow[k];        // dims 0..31
      b1[e] = (__bf16)qrow[32 + k];   // dims 32..63
    }
  }

  float ts[TOPK]; int ti[TOPK];
#pragma unroll
  for (int i = 0; i < TOPK; ++i) { ts[i] = NEG_INF; ti[i] = 0; }

  int nbeg = blockIdx.x * chunk;
  int nend = nbeg + chunk;
  if (nend > N) nend = N;

  for (int nbase = nbeg; nbase < nend; nbase += TILE) {
    // --- cooperative load + dequant of 64x64 tile into LDS (bf16) ---------
#pragma unroll
    for (int rg = 0; rg < 4; ++rg) {
      int f = (t + 128 * rg) * 8;    // 4 rounds x 1024 floats = 4096
      int row = f >> 6, col = f & 63;
      int g = nbase + row;
      unsigned* dst = (unsigned*)tileB + (f >> 1);
      if (g < N) {
        const float* src = mem + g * 64 + col;
        v4f x0 = *(const v4f*)src;
        v4f x1 = *(const v4f*)(src + 4);
        float v[8];
#pragma unroll
        for (int i = 0; i < 4; ++i)
          v[i]     = fmaf(rintf(fmaf(x0[i], invs, zp)), sczp[0], c1);
#pragma unroll
        for (int i = 0; i < 4; ++i)
          v[4 + i] = fmaf(rintf(fmaf(x1[i], invs, zp)), sczp[0], c1);
#pragma unroll
        for (int i = 0; i < 4; ++i) {
          unsigned short lo = __builtin_bit_cast(unsigned short, (__bf16)v[2 * i]);
          unsigned short hh = __builtin_bit_cast(unsigned short, (__bf16)v[2 * i + 1]);
          dst[i] = (unsigned)lo | ((unsigned)hh << 16);
        }
        if (rg == 0 && g + TILE < N)
          __builtin_prefetch(src + TILE * 64, 0, 3);   // next macro-tile
      } else {
#pragma unroll
        for (int i = 0; i < 4; ++i) dst[i] = 0u;
      }
    }
    if (t < TILE) {
      int g = nbase + t;
      s_aw[t] = (g < N) ? attw[g] : 0.0f;
    }
    __syncthreads();

    // --- 4 sub-tiles of 16 mem rows -> 8 WMMAs per barrier pair -----------
#pragma unroll
    for (int st = 0; st < 4; ++st) {
      int sb = st * 16;
      // A fragments = mem rows (contiguous k order, same as B:
      // lane m = lane&15; element e -> k=16*hi+e)
      const __bf16* arow = tileB + (sb + nq) * 64;
      v8bf a0l = *(const v8bf*)(arow + 16 * hi);
      v8bf a0h = *(const v8bf*)(arow + 8 + 16 * hi);
      v8bf a1l = *(const v8bf*)(arow + 32 + 16 * hi);
      v8bf a1h = *(const v8bf*)(arow + 40 + 16 * hi);
      v16bf a0 = __builtin_shufflevector(a0l, a0h, 0,1,2,3,4,5,6,7,8,9,10,11,12,13,14,15);
      v16bf a1 = __builtin_shufflevector(a1l, a1h, 0,1,2,3,4,5,6,7,8,9,10,11,12,13,14,15);

      // bias for rows sb+8*hi .. sb+8*hi+7 : two 16B LDS loads
      v4f bw0 = *(const v4f*)(s_aw + sb + 8 * hi);
      v4f bw1 = *(const v4f*)(s_aw + sb + 8 * hi + 4);

      v8f acc = {};
      acc = __builtin_amdgcn_wmma_f32_16x16x32_bf16(false, a0, false, b0,
                                                    (short)0, acc, false, false);
      acc = __builtin_amdgcn_wmma_f32_16x16x32_bf16(false, a1, false, b1,
                                                    (short)0, acc, false, false);

      // acc[r] = score(mem row sb + r + 8*hi, query nq)
      float sv[8];
#pragma unroll
      for (int r = 0; r < 4; ++r) sv[r]     = fmaf(acc[r],     0.125f, bw0[r]);
#pragma unroll
      for (int r = 0; r < 4; ++r) sv[4 + r] = fmaf(acc[4 + r], 0.125f, bw1[r]);

      // fast gate: only enter insertion path if some score beats 5th-best
      float mx = fmaxf(fmaxf(fmaxf(sv[0], sv[1]), fmaxf(sv[2], sv[3])),
                       fmaxf(fmaxf(sv[4], sv[5]), fmaxf(sv[6], sv[7])));
      if (mx > ts[TOPK - 1]) {
        int base = nbase + sb + 8 * hi;
#pragma unroll
        for (int r = 0; r < 8; ++r) {
          float s = sv[r];
          int nn = base + r;
          if (nn < N && s > ts[TOPK - 1]) {
            int p = TOPK - 1;
#pragma unroll
            for (int qq = TOPK - 1; qq > 0; --qq) {
              if (s > ts[qq - 1]) { ts[qq] = ts[qq - 1]; ti[qq] = ti[qq - 1]; p = qq - 1; }
            }
            ts[p] = s; ti[p] = nn;
          }
        }
      }
    }
    __syncthreads();  // tileB reuse next iteration
  }

  // --- merge the two half-wave partial lists per query via shuffles -------
#pragma unroll
  for (int i = 0; i < TOPK; ++i) {
    float os = __shfl(ts[i], lane + 16);  // partner's i-th best
    int   oi = __shfl(ti[i], lane + 16);
    if (lane < 16 && os > ts[TOPK - 1]) {
      int p = TOPK - 1;
#pragma unroll
      for (int qq = TOPK - 1; qq > 0; --qq) {
        if (os > ts[qq - 1]) { ts[qq] = ts[qq - 1]; ti[qq] = ti[qq - 1]; p = qq - 1; }
      }
      ts[p] = os; ti[p] = oi;
    }
  }

  if (lane < 16) {
    int q = w * 16 + lane;
    float* ps = pscore + (blockIdx.x * 64 + q) * TOPK;
    int*   pi = pidx   + (blockIdx.x * 64 + q) * TOPK;
#pragma unroll
    for (int i = 0; i < TOPK; ++i) { ps[i] = ts[i]; pi[i] = ti[i]; }
  }
}

// ---------------------------------------------------------------------------
// Kernel D: per query, merge CB partial top-5 lists -> global top-5,
//           softmax, gather+dequant 5 rows, h = sum dist*row, out = h @ Wv.T
// ---------------------------------------------------------------------------
__global__ __launch_bounds__(128) void final_kernel(
    const float* __restrict__ pscore, const int* __restrict__ pidx,
    const float* __restrict__ mem, const float* __restrict__ Wv,
    const float* __restrict__ sczp, float* __restrict__ out, int nblocks) {
  __shared__ float cs[128 * TOPK];
  __shared__ int   ci[128 * TOPK];
  __shared__ float dist[TOPK];
  __shared__ int   gidx[TOPK];
  __shared__ float h[64];

  int q = blockIdx.x;
  int t = threadIdx.x;

  float ts[TOPK]; int ti[TOPK];
#pragma unroll
  for (int i = 0; i < TOPK; ++i) { ts[i] = NEG_INF; ti[i] = 0; }

  int total = nblocks * TOPK;
  for (int c = t; c < total; c += 128) {
    int blk = c / TOPK, i = c % TOPK;
    float s = pscore[(blk * 64 + q) * TOPK + i];
    int   n = pidx  [(blk * 64 + q) * TOPK + i];
    if (s > ts[TOPK - 1]) {
      int p = TOPK - 1;
#pragma unroll
      for (int qq = TOPK - 1; qq > 0; --qq) {
        if (s > ts[qq - 1]) { ts[qq] = ts[qq - 1]; ti[qq] = ti[qq - 1]; p = qq - 1; }
      }
      ts[p] = s; ti[p] = n;
    }
  }
#pragma unroll
  for (int i = 0; i < TOPK; ++i) { cs[t * TOPK + i] = ts[i]; ci[t * TOPK + i] = ti[i]; }
  __syncthreads();

  if (t == 0) {
    float bs[TOPK]; int bi[TOPK];
#pragma unroll
    for (int i = 0; i < TOPK; ++i) { bs[i] = NEG_INF; bi[i] = 0; }
    for (int c = 0; c < 128 * TOPK; ++c) {
      float s = cs[c]; int n = ci[c];
      if (s > bs[TOPK - 1]) {
        int p = TOPK - 1;
#pragma unroll
        for (int qq = TOPK - 1; qq > 0; --qq) {
          if (s > bs[qq - 1]) { bs[qq] = bs[qq - 1]; bi[qq] = bi[qq - 1]; p = qq - 1; }
        }
        bs[p] = s; bi[p] = n;
      }
    }
    float mx = bs[0];
    float e[TOPK], sum = 0.f;
#pragma unroll
    for (int i = 0; i < TOPK; ++i) { e[i] = expf(bs[i] - mx); sum += e[i]; }
#pragma unroll
    for (int i = 0; i < TOPK; ++i) { dist[i] = e[i] / sum; gidx[i] = bi[i]; }
  }
  __syncthreads();

  float scale = sczp[0], zp = sczp[1], invs = sczp[2];
  if (t < 64) {
    float acc = 0.f;
#pragma unroll
    for (int i = 0; i < TOPK; ++i) {
      float x = mem[gidx[i] * 64 + t];
      acc += dist[i] * ((rintf(x * invs + zp) - zp) * scale);
    }
    h[t] = acc;
  }
  __syncthreads();
  if (t < 64) {
    float acc = 0.f;
    for (int d = 0; d < 64; ++d) acc += h[d] * Wv[t * 64 + d];
    out[q * 64 + t] = acc;
  }
}

// ---------------------------------------------------------------------------
extern "C" void kernel_launch(void* const* d_in, const int* in_sizes, int n_in,
                              void* d_out, int out_size, void* d_ws, size_t ws_size,
                              hipStream_t stream) {
  const float* query = (const float*)d_in[0];
  const float* mem   = (const float*)d_in[1];
  const float* attw  = (const float*)d_in[2];
  const float* Wq    = (const float*)d_in[3];
  const float* Wk    = (const float*)d_in[4];
  const float* Wv    = (const float*)d_in[5];
  (void)n_in; (void)out_size; (void)ws_size;

  int total = in_sizes[1];     // N * 64
  int N = total / 64;

  float* ws     = (float*)d_ws;
  float* sczp   = ws;                       // [0]=scale [1]=zp [2]=1/scale [3]=-zp*scale
  float* q2f    = ws + 16;                  // 4096 floats
  float* mmpart = q2f + 4096;               // 2*MMB floats
  float* pscore = mmpart + 2 * MMB;         // CB*64*TOPK floats
  int*   pidx   = (int*)(pscore + CB * 64 * TOPK);

  int chunk = ((N + CB - 1) / CB + TILE - 1) & ~(TILE - 1);  // TILE-aligned

  mm_kernel<<<MMB, 256, 0, stream>>>(mem, total, mmpart);
  prep_kernel<<<1, 256, 0, stream>>>(mmpart, MMB, query, Wq, Wk, sczp, q2f);
  score_topk_kernel<<<CB, 128, 0, stream>>>(mem, attw, sczp, q2f, pscore, pidx, N, chunk);
  final_kernel<<<64, 128, 0, stream>>>(pscore, pidx, mem, Wv, sczp, (float*)d_out, CB);
}
